// RecurrentModel_4990751998447
// MI455X (gfx1250) — matrix operand order repaired and checked by reading, not compiled, not run on hardware
//
#include <hip/hip_runtime.h>
#include <hip/hip_bf16.h>

// Sizes from the reference
#define B_   8192
#define STIM 255
#define IN_  256
#define H_   1024
#define OUT_ 2

typedef _Float16 h16;
typedef __attribute__((ext_vector_type(2)))  float    v2f;
typedef __attribute__((ext_vector_type(8)))  float    v8f;
typedef __attribute__((ext_vector_type(16))) _Float16 v16h;

union V16 { v16h v; uint4 q[2]; };

#define WMMA_F16(A, B, C) \
    __builtin_amdgcn_wmma_f32_16x16x32_f16(false, (A), false, (B), (short)0, (C), false, false)

// ---------------------------------------------------------------------------
// Prep: resolve concat(stimulus, reward) -> X and split X, W_ih into f16
// hi/lo pairs (x = hi + lo, ~22 effective mantissa bits).
// ws layout (h16 elements): Xhi[2M] | Xlo[2M] | Whi[768K] | Wlo[768K]
// ---------------------------------------------------------------------------
#define XN_  (B_ * IN_)          // 2,097,152
#define WN_  (3 * H_ * IN_)      // 786,432

__global__ __launch_bounds__(256)
void split_f16_kernel(const float* __restrict__ stim,
                      const float* __restrict__ reward,
                      const float* __restrict__ Wih,
                      h16* __restrict__ xhi, h16* __restrict__ xlo,
                      h16* __restrict__ whi, h16* __restrict__ wlo)
{
    for (int idx = blockIdx.x * blockDim.x + threadIdx.x; idx < XN_ + WN_;
         idx += gridDim.x * blockDim.x) {
        float x; h16* dhi; h16* dlo; int o;
        if (idx < XN_) {
            const int b = idx >> 8, k = idx & (IN_ - 1);
            x = (k < STIM) ? stim[(size_t)b * STIM + k] : reward[b];
            dhi = xhi; dlo = xlo; o = idx;
        } else {
            o = idx - XN_;
            x = Wih[o];
            dhi = whi; dlo = wlo;
        }
        const h16 hi = (h16)x;                 // RTNE
        const h16 lo = (h16)(x - (float)hi);   // residual
        dhi[o] = hi; dlo[o] = lo;
    }
}

// ---------------------------------------------------------------------------
// GEMM + GRU gates via split-f16 WMMA: gi = X @ W_ih^T, then
// r=sig(gi_r+bhh_r), z=sig(gi_z+bhh_z), n=tanh(gi_n+r*bhh_n), h=(1-z)*n.
// One wave -> one 16(M) x 16(N) h tile; K=256 in 8 steps of 32.
// A: 16 contiguous K per lane at k0+16h, M = lane%16 (walks X rows), same K order as B;
// B: 16 contiguous K per lane at k0+16h, N = lane%16 (walks W_ih rows).
// ---------------------------------------------------------------------------
__global__ __launch_bounds__(256)
void gru_wmma_f16x3_kernel(const h16* __restrict__ xhi, const h16* __restrict__ xlo,
                           const h16* __restrict__ whi, const h16* __restrict__ wlo,
                           const float* __restrict__ bih, const float* __restrict__ bhh,
                           float* __restrict__ core_out, float* __restrict__ core_hid)
{
    const int lane = threadIdx.x & 31;
    const int wave = threadIdx.x >> 5;
    const int half = lane >> 4;
    const int lr   = lane & 15;

    const int m0 = blockIdx.x * 16;
    const int n0 = (blockIdx.y * 8 + wave) * 16;

    const size_t xoff = (size_t)(m0 + lr) * IN_;
    const size_t woff = (size_t)(n0 + lr) * IN_;   // gate r row; +H_*IN_ for z, +2*H_*IN_ for n
    const int aoff = 16 * half;
    const int boff = 16 * half;

    v8f accr = {}; v8f accz = {}; v8f accn = {};

#pragma unroll 2
    for (int k0 = 0; k0 < IN_; k0 += 32) {
        V16 Ahi, Alo;
        Ahi.q[0] = *(const uint4*)(xhi + xoff + k0 + aoff);
        Ahi.q[1] = *(const uint4*)(xhi + xoff + k0 + aoff + 8);
        Alo.q[0] = *(const uint4*)(xlo + xoff + k0 + aoff);
        Alo.q[1] = *(const uint4*)(xlo + xoff + k0 + aoff + 8);

        {   // gate r
            V16 Bhi, Blo;
            Bhi.q[0] = *(const uint4*)(whi + woff + k0 + boff);
            Bhi.q[1] = *(const uint4*)(whi + woff + k0 + boff + 8);
            Blo.q[0] = *(const uint4*)(wlo + woff + k0 + boff);
            Blo.q[1] = *(const uint4*)(wlo + woff + k0 + boff + 8);
            accr = WMMA_F16(Ahi.v, Bhi.v, accr);
            accr = WMMA_F16(Ahi.v, Blo.v, accr);
            accr = WMMA_F16(Alo.v, Bhi.v, accr);
        }
        {   // gate z
            const size_t wz = woff + (size_t)H_ * IN_;
            V16 Bhi, Blo;
            Bhi.q[0] = *(const uint4*)(whi + wz + k0 + boff);
            Bhi.q[1] = *(const uint4*)(whi + wz + k0 + boff + 8);
            Blo.q[0] = *(const uint4*)(wlo + wz + k0 + boff);
            Blo.q[1] = *(const uint4*)(wlo + wz + k0 + boff + 8);
            accz = WMMA_F16(Ahi.v, Bhi.v, accz);
            accz = WMMA_F16(Ahi.v, Blo.v, accz);
            accz = WMMA_F16(Alo.v, Bhi.v, accz);
        }
        {   // gate n
            const size_t wn = woff + (size_t)(2 * H_) * IN_;
            V16 Bhi, Blo;
            Bhi.q[0] = *(const uint4*)(whi + wn + k0 + boff);
            Bhi.q[1] = *(const uint4*)(whi + wn + k0 + boff + 8);
            Blo.q[0] = *(const uint4*)(wlo + wn + k0 + boff);
            Blo.q[1] = *(const uint4*)(wlo + wn + k0 + boff + 8);
            accn = WMMA_F16(Ahi.v, Bhi.v, accn);
            accn = WMMA_F16(Ahi.v, Blo.v, accn);
            accn = WMMA_F16(Alo.v, Bhi.v, accn);
        }
    }

    // C/D: VGPR i, lane L -> M = i + (L>=16 ? 8 : 0), N = L%16 (per-lane const col)
    const int col = n0 + lr;
    const float bir = bih[col], biz = bih[col + H_], bin = bih[col + 2 * H_];
    const float bhr = bhh[col], bhz = bhh[col + H_], bhn = bhh[col + 2 * H_];

#pragma unroll
    for (int i = 0; i < 8; ++i) {
        const int m = m0 + i + half * 8;
        const float gr = accr[i] + bir + bhr;            // gh = bias_hh (h0 == 0)
        const float gz = accz[i] + biz + bhz;
        const float r = 1.0f / (1.0f + __expf(-gr));
        const float z = 1.0f / (1.0f + __expf(-gz));
        const float gn = accn[i] + bin + r * bhn;
        const float e = __expf(-2.0f * gn);              // tanh(gn)
        const float n = (1.0f - e) / (1.0f + e);
        const float h = (1.0f - z) * n;
        core_out[(size_t)m * H_ + col] = h;
        core_hid[(size_t)m * H_ + col] = h;
    }
}

// ---------------------------------------------------------------------------
// Fallback (no ws dependence): exact f32 WMMA 16x16x4 version from round 1.
// ---------------------------------------------------------------------------
__global__ __launch_bounds__(256)
void gru_wmma_f32_kernel(const float* __restrict__ stim,
                         const float* __restrict__ reward,
                         const float* __restrict__ Wih,
                         const float* __restrict__ bih,
                         const float* __restrict__ bhh,
                         float* __restrict__ core_out,
                         float* __restrict__ core_hid)
{
    const int lane = threadIdx.x & 31;
    const int wave = threadIdx.x >> 5;
    const int half = lane >> 4;
    const int lr   = lane & 15;
    const int m0 = blockIdx.x * 16;
    const int n0 = (blockIdx.y * 8 + wave) * 16;
    const int koff = half * 2;

    const int xrow_i = m0 + lr;
    const float* __restrict__ xrow = stim + (size_t)xrow_i * STIM;
    const float* __restrict__ wr = Wih + (size_t)(n0 + lr) * IN_;
    const float* __restrict__ wz = wr + (size_t)H_ * IN_;
    const float* __restrict__ wn = wr + (size_t)(2 * H_) * IN_;

    v8f accr = {}; v8f accz = {}; v8f accn = {};
#pragma unroll 4
    for (int k0 = 0; k0 < IN_; k0 += 4) {
        const int k = k0 + koff;
        v2f a;
        a.x = xrow[k];
        a.y = (k == STIM - 1) ? reward[xrow_i] : xrow[k + 1];
        v2f br = *(const v2f*)(wr + k);
        v2f bz = *(const v2f*)(wz + k);
        v2f bn = *(const v2f*)(wn + k);
        accr = __builtin_amdgcn_wmma_f32_16x16x4_f32(false, a, false, br, (short)0, accr, false, false);
        accz = __builtin_amdgcn_wmma_f32_16x16x4_f32(false, a, false, bz, (short)0, accz, false, false);
        accn = __builtin_amdgcn_wmma_f32_16x16x4_f32(false, a, false, bn, (short)0, accn, false, false);
    }

    const int col = n0 + lr;
    const float bir = bih[col], biz = bih[col + H_], bin = bih[col + 2 * H_];
    const float bhr = bhh[col], bhz = bhh[col + H_], bhn = bhh[col + 2 * H_];
#pragma unroll
    for (int i = 0; i < 8; ++i) {
        const int m = m0 + i + half * 8;
        const float gr = accr[i] + bir + bhr;
        const float gz = accz[i] + biz + bhz;
        const float r = 1.0f / (1.0f + __expf(-gr));
        const float z = 1.0f / (1.0f + __expf(-gz));
        const float gn = accn[i] + bin + r * bhn;
        const float e = __expf(-2.0f * gn);
        const float n = (1.0f - e) / (1.0f + e);
        const float h = (1.0f - z) * n;
        core_out[(size_t)m * H_ + col] = h;
        core_hid[(size_t)m * H_ + col] = h;
    }
}

// ---------------------------------------------------------------------------
// Readout: linear = h @ w_readout^T + b_readout (OUT=2), then sigmoid.
// One wave32 per batch row; h is L2-resident from the GEMM kernel.
// ---------------------------------------------------------------------------
__global__ __launch_bounds__(256)
void readout_kernel(const float* __restrict__ h,
                    const float* __restrict__ wro,
                    const float* __restrict__ bro,
                    float* __restrict__ lin,
                    float* __restrict__ sig)
{
    const int lane = threadIdx.x & 31;
    const int wave = threadIdx.x >> 5;
    const int b = blockIdx.x * 8 + wave;
    const float* __restrict__ hrow = h + (size_t)b * H_;

    float s0 = 0.0f, s1 = 0.0f;
#pragma unroll 4
    for (int j = lane; j < H_; j += 32) {
        const float hv = hrow[j];
        s0 += hv * wro[j];
        s1 += hv * wro[H_ + j];
    }
#pragma unroll
    for (int off = 16; off > 0; off >>= 1) {
        s0 += __shfl_down(s0, off, 32);
        s1 += __shfl_down(s1, off, 32);
    }
    if (lane == 0) {
        const float l0 = s0 + bro[0];
        const float l1 = s1 + bro[1];
        lin[(size_t)b * OUT_ + 0] = l0;
        lin[(size_t)b * OUT_ + 1] = l1;
        sig[(size_t)b * OUT_ + 0] = 1.0f / (1.0f + __expf(-l0));
        sig[(size_t)b * OUT_ + 1] = 1.0f / (1.0f + __expf(-l1));
    }
}

extern "C" void kernel_launch(void* const* d_in, const int* in_sizes, int n_in,
                              void* d_out, int out_size, void* d_ws, size_t ws_size,
                              hipStream_t stream) {
    const float* stim   = (const float*)d_in[0];   // (8192,1,255)
    const float* reward = (const float*)d_in[1];   // (8192,1)
    const float* Wih    = (const float*)d_in[2];   // (3072,256)
    // d_in[3] = weight_hh is dead: h0 == 0 -> gh = bias_hh
    const float* bih    = (const float*)d_in[4];   // (3072,)
    const float* bhh    = (const float*)d_in[5];   // (3072,)
    const float* wro    = (const float*)d_in[6];   // (2,1024)
    const float* bro    = (const float*)d_in[7];   // (2,)

    float* out      = (float*)d_out;
    float* core_out = out;                                   // (8192,1,1024)
    float* core_hid = core_out + (size_t)B_ * H_;            // (8192,1,1024)
    float* lin      = core_hid + (size_t)B_ * H_;            // (8192,1,2)
    float* sig      = lin + (size_t)B_ * OUT_;               // (8192,1,2)

    const dim3 grid(B_ / 16, (H_ / 16) / 8);   // (512, 8); 8 waves/block

    const size_t ws_needed = (size_t)(2 * XN_ + 2 * WN_) * sizeof(h16);  // ~11.5 MB
    if (ws_size >= ws_needed) {
        h16* xhi = (h16*)d_ws;
        h16* xlo = xhi + XN_;
        h16* whi = xlo + XN_;
        h16* wlo = whi + WN_;
        split_f16_kernel<<<4096, 256, 0, stream>>>(stim, reward, Wih, xhi, xlo, whi, wlo);
        gru_wmma_f16x3_kernel<<<grid, 256, 0, stream>>>(xhi, xlo, whi, wlo, bih, bhh,
                                                        core_out, core_hid);
    } else {
        gru_wmma_f32_kernel<<<grid, 256, 0, stream>>>(stim, reward, Wih, bih, bhh,
                                                      core_out, core_hid);
    }
    readout_kernel<<<B_ / 8, 256, 0, stream>>>(core_out, wro, bro, lin, sig);
}
